// MaskedConv2d_58437325030066
// MI455X (gfx1250) — matrix operand order repaired and checked by reading, not compiled, not run on hardware
//
#include <hip/hip_runtime.h>
#include <hip/hip_bf16.h>

typedef __bf16 v16bf __attribute__((ext_vector_type(16)));
typedef float  v8f   __attribute__((ext_vector_type(8)));
typedef unsigned int u32x4 __attribute__((ext_vector_type(4)));
typedef unsigned int tdm_g0_t __attribute__((ext_vector_type(4)));
typedef int          tdm_i4_t __attribute__((ext_vector_type(4)));
typedef int          tdm_i8_t __attribute__((ext_vector_type(8)));

#define CIN   128
#define COUT  256
#define HW    56
#define SP    3136          // 56*56
#define KDIM  1152          // 128*9
#define NIMG  32
#define NDIM  (NIMG * SP)   // 100352

#define BM 128
#define BN 128
#define BK 32
#define LDSROW 40           // 32 k-values + 8 pad (ushorts) -> 80B row stride (16B aligned)

#if __has_builtin(__builtin_amdgcn_tensor_load_to_lds) && \
    __has_builtin(__builtin_amdgcn_s_wait_tensorcnt)
#define USE_TDM 1
#else
#define USE_TDM 0
#endif

__device__ __forceinline__ unsigned short f32_to_bf16(float f) {
    unsigned int u = __float_as_uint(f);
    u += 0x7FFFu + ((u >> 16) & 1u);   // round-to-nearest-even
    return (unsigned short)(u >> 16);
}

// ---- Kernel 1: wbf[oc][k] = bf16(weight[oc][k] * mask[k]),  k = c*9 + kh*3 + kw ----
__global__ void mask_weight_bf16(const float* __restrict__ w,
                                 const float* __restrict__ mask,
                                 unsigned short* __restrict__ wbf) {
    int idx = blockIdx.x * 256 + threadIdx.x;
    if (idx >= COUT * KDIM) return;
    int k = idx % KDIM;                 // mask is [C][3][3] flat == k
    wbf[idx] = f32_to_bf16(w[idx] * mask[k]);
}

// ---- Kernel 2: implicit-GEMM conv with V_WMMA_F32_16X16X32_BF16 + TDM A-tile ----
__global__ void __launch_bounds__(256)
masked_conv_wmma(const float* __restrict__ x,
                 const unsigned short* __restrict__ wbf,
                 float* __restrict__ out) {
    __shared__ __align__(16) unsigned short ldsA[BM * LDSROW];  // [m][k], 80B row stride
    __shared__ __align__(16) unsigned short ldsB[BN * LDSROW];  // [n][k], 80B row stride

    const int tid   = threadIdx.x;
    const int lane  = tid & 31;
    const int wave  = tid >> 5;
    const int waveM = wave & 3;   // 4 waves along M -> 32 rows each
    const int waveN = wave >> 2;  // 2 waves along N -> 64 cols each

    const int mBase = blockIdx.y * BM;
    const int nBase = blockIdx.x * BN;

    // Per-thread cooperative-load coordinates (thread t -> tile row/col t/2, k-half t%2)
    const int rcLocal = tid >> 1;            // 0..127
    const int kHalf   = (tid & 1) << 4;      // 0 or 16 (ushort offset within BK)

    // Decode this thread's im2col column once
    const int gcol = nBase + rcLocal;
    const int nImg = gcol / SP;
    const int sp   = gcol - nImg * SP;
    const int oh   = sp / HW;
    const int ow   = sp - oh * HW;
    const float* xbase = x + (size_t)nImg * CIN * SP;

#if USE_TDM
    // TDM descriptor group1 (constant): workgroup_mask=0, data_size=1 (2B),
    // pad_enable=1, pad_interval=3 (16 DWORDs = one 64B tile row),
    // pad_amount=3 (4 DWORDs = 16B) -> LDS row stride 80B, matching LDSROW.
    // tensor_dim0=1152, tensor_dim1=256, tile_dim0=32, tile_dim1=128,
    // tensor_dim0_stride=1152.
    const tdm_i8_t tdmG1 = {
        (int)0x06D10000,                    // mask/data_size/pad fields
        (int)((KDIM & 0xFFFF) << 16),       // abar=0 | tensor_dim0[15:0]
        (int)((KDIM >> 16) | ((COUT & 0xFFFF) << 16)),  // dim0 hi | tensor_dim1 lo
        (int)((COUT >> 16) | (BK << 16)),   // tensor_dim1 hi | tile_dim0
        (int)(BM),                          // tile_dim1=128 | tile_dim2=0
        (int)(KDIM),                        // tensor_dim0_stride[31:0]
        0, 0                                // stride hi / tensor_dim1_stride
    };
    const tdm_i4_t tdmZ4 = {};
#if __clang_major__ >= 23
    const tdm_i8_t tdmZ8 = {};
#endif
    const unsigned ldsAaddr = (unsigned)(size_t)(&ldsA[0]);
    const unsigned long long wbfAddr =
        (unsigned long long)wbf + (unsigned long long)mBase * (KDIM * 2ull);
#endif

    v8f acc[2][4];
    v8f zero = {};
    #pragma unroll
    for (int i = 0; i < 2; ++i)
        #pragma unroll
        for (int j = 0; j < 4; ++j)
            acc[i][j] = zero;

    for (int kBase = 0; kBase < KDIM; kBase += BK) {
        // ---- A tile: 128 x 32 bf16 masked weights ----
#if USE_TDM
        if (wave == 0) {
            // group0: count=1 | lds_addr | global_addr (57b) | type=2
            unsigned long long ga = wbfAddr + (unsigned long long)kBase * 2ull;
            tdm_g0_t g0 = { 1u, ldsAaddr, (unsigned)ga,
                            ((unsigned)(ga >> 32) & 0x01FFFFFFu) | 0x80000000u };
#if __clang_major__ >= 23
            __builtin_amdgcn_tensor_load_to_lds(g0, tdmG1, tdmZ4, tdmZ4, tdmZ8, 0);
#else
            __builtin_amdgcn_tensor_load_to_lds(g0, tdmG1, tdmZ4, tdmZ4, 0);
#endif
        }
#else
        {
            const unsigned short* grow =
                wbf + (size_t)(mBase + rcLocal) * KDIM + kBase + kHalf;
            u32x4 q0 = ((const u32x4*)grow)[0];
            u32x4 q1 = ((const u32x4*)grow)[1];
            u32x4* lp = (u32x4*)&ldsA[rcLocal * LDSROW + kHalf];
            lp[0] = q0;
            lp[1] = q1;
        }
#endif
        // ---- B tile: im2col gather, 128 cols x 32 k, stored [col][k] ----
        // Runs concurrently with the TDM copy above.
        {
            const int c0 = (kBase + kHalf) / 9;
            if (c0 + 4 < CIN)   // warm L2 for next k-step's x region
                __builtin_prefetch(xbase + (size_t)(c0 + 4) * SP + oh * HW + ow, 0, 1);

            unsigned int packed[8];
            #pragma unroll
            for (int p = 0; p < 8; ++p) {
                unsigned short h[2];
                #pragma unroll
                for (int e = 0; e < 2; ++e) {
                    int gk  = kBase + kHalf + 2 * p + e;
                    int c   = gk / 9;
                    int rem = gk - c * 9;
                    int kh  = rem / 3;
                    int kw  = rem - kh * 3;
                    int ih  = oh + kh - 1;
                    int iw  = ow + kw - 1;
                    float v = 0.0f;
                    if ((unsigned)ih < HW && (unsigned)iw < HW)
                        v = xbase[(size_t)c * SP + ih * HW + iw];
                    h[e] = f32_to_bf16(v);
                }
                packed[p] = (unsigned)h[0] | ((unsigned)h[1] << 16);
            }
            u32x4 q0 = {packed[0], packed[1], packed[2], packed[3]};
            u32x4 q1 = {packed[4], packed[5], packed[6], packed[7]};
            u32x4* lp = (u32x4*)&ldsB[rcLocal * LDSROW + kHalf];
            lp[0] = q0;
            lp[1] = q1;
        }
#if USE_TDM
        if (wave == 0)
            __builtin_amdgcn_s_wait_tensorcnt(0);   // A tile landed in LDS
#endif
        __syncthreads();

        // ---- build fragments per ISA 7.12.2 lane layouts, then 8 WMMAs ----
        union FragU { v16bf v; u32x4 q[2]; };
        FragU afrag[2], bfrag[4];

        const int aSel = (lane < 16) ? 0 : 8;    // A: K 0-7/16-23 vs 8-15/24-31
        #pragma unroll
        for (int i = 0; i < 2; ++i) {
            int row = waveM * 32 + i * 16 + (lane & 15);
            const unsigned short* base = &ldsA[row * LDSROW];
            afrag[i].q[0] = *(const u32x4*)(base + aSel);
            afrag[i].q[1] = *(const u32x4*)(base + aSel + 16);
        }
        const int bSel = (lane < 16) ? 0 : 8;    // B: K 0-7/16-23 vs 8-15/24-31
        #pragma unroll
        for (int j = 0; j < 4; ++j) {
            int col = waveN * 64 + j * 16 + (lane & 15);
            const unsigned short* base = &ldsB[col * LDSROW];
            bfrag[j].q[0] = *(const u32x4*)(base + bSel);
            bfrag[j].q[1] = *(const u32x4*)(base + bSel + 16);
        }

        #pragma unroll
        for (int i = 0; i < 2; ++i)
            #pragma unroll
            for (int j = 0; j < 4; ++j)
                acc[i][j] = __builtin_amdgcn_wmma_f32_16x16x32_bf16(
                    false, afrag[i].v, false, bfrag[j].v,
                    (short)0, acc[i][j], false, false);

        __syncthreads();
    }

    // ---- store D: VGPR r -> M = r + (lane>=16 ? 8 : 0), N = lane&15 ----
    const int nColBase = nBase + waveN * 64;
    #pragma unroll
    for (int j = 0; j < 4; ++j) {
        int gc = nColBase + j * 16 + (lane & 15);
        int ni = gc / SP;
        int s  = gc - ni * SP;
        float* obase = out + (size_t)ni * COUT * SP + s;
        #pragma unroll
        for (int i = 0; i < 2; ++i) {
            int ocBase = mBase + waveM * 32 + i * 16 + ((lane >> 4) << 3);
            #pragma unroll
            for (int r = 0; r < 8; ++r)
                obase[(size_t)(ocBase + r) * SP] = acc[i][j][r];
        }
    }
}

extern "C" void kernel_launch(void* const* d_in, const int* in_sizes, int n_in,
                              void* d_out, int out_size, void* d_ws, size_t ws_size,
                              hipStream_t stream) {
    const float* x    = (const float*)d_in[0];   // [32,128,56,56]
    const float* w    = (const float*)d_in[1];   // [256,128,3,3]
    const float* mask = (const float*)d_in[2];   // [128,3,3]
    unsigned short* wbf = (unsigned short*)d_ws; // 256*1152 bf16 = 576KB scratch
    float* out = (float*)d_out;                  // [32,256,56,56]

    mask_weight_bf16<<<(COUT * KDIM + 255) / 256, 256, 0, stream>>>(w, mask, wbf);

    dim3 grid(NDIM / BN, COUT / BM);             // 784 x 2
    masked_conv_wmma<<<grid, dim3(256), 0, stream>>>(x, wbf, out);
}
